// BaselineAttention_25984552141259
// MI455X (gfx1250) — matrix operand order repaired and checked by reading, not compiled, not run on hardware
//
#include <hip/hip_runtime.h>

// Flash-attention forward for QKV[3][2*8][4096][64] fp32 on gfx1250.
//
// Pass 1 (fa_pack_kernel): convert K (A-frag layout) and V (B-frag layout)
// to f16, interleaved per 32-row KV block: 8 fragments x 32 lanes x 32B
// = 8KB per block, contiguous.
//
// Pass 2 (fa_fwd_kernel): workgroup = 8 waves sharing one head (bh); the
// packed 8KB KV block is staged into LDS ONCE per workgroup (TDM
// tensor_load_to_lds, double-buffered) instead of 8x from L2. Each wave32
// owns TWO 16-row Q tiles: 8 LDS fragment reads feed 16
// v_wmma_f32_16x16x32_f16 per iteration. S is computed transposed (K*Q^T)
// so P lands directly in A-fragment layout for the P*V WMMA.

typedef __attribute__((ext_vector_type(16))) _Float16 v16h;
typedef __attribute__((ext_vector_type(8)))  float    v8f;
typedef __attribute__((ext_vector_type(4)))  float    v4f;
typedef __attribute__((ext_vector_type(4)))  unsigned int u32x4;
typedef __attribute__((ext_vector_type(8)))  int      i32x8;
typedef __attribute__((ext_vector_type(4)))  int      i32x4;

#define ATTN_N  4096
#define ATTN_D  64
#define ATTN_BH 16
#define ATTN_JB (ATTN_N / 32)      // 128 KV blocks per head
#define BLK_HALVES 4096            // 8 frags * 32 lanes * 16 halves (8KB)

#if defined(__AMDGCN__) && __has_builtin(__builtin_amdgcn_tensor_load_to_lds) && \
    __has_builtin(__builtin_amdgcn_s_wait_tensorcnt)
#define FA_HAVE_TDM 1
#else
#define FA_HAVE_TDM 0
#endif

static __device__ __forceinline__ v8f wmma_f16(v16h a, v16h b, v8f c) {
  return __builtin_amdgcn_wmma_f32_16x16x32_f16(false, a, false, b, (short)0, c,
                                                false, false);
}

// ---------------------------------------------------------------------------
// Stage one 8KB packed KV block from global into LDS.
// TDM path: wave 0 issues one tensor_load_to_lds with a 1-D D# descriptor
// (CDNA5 ISA): count=1, data_size=8B,
// tile_dim0 = tensor_dim0 = tensor_dim0_stride = 1024 elements (= 8192 B).
// Fallback: cooperative 256-thread register copy (32B per thread).
// ---------------------------------------------------------------------------
static __device__ __forceinline__ void fa_stage(_Float16* dstLds,
                                                const _Float16* gsrc,
                                                int tid, int wave) {
#if FA_HAVE_TDM
  if (wave == 0) {
    const unsigned long long ga = (unsigned long long)(size_t)gsrc;
    u32x4 g0 = {};
    g0[0] = 1u;                                   // count = 1 valid descriptor
    g0[1] = (unsigned)(size_t)dstLds;             // lds_addr (low 32b of flat)
    g0[2] = (unsigned)ga;                         // global_addr[31:0]
    g0[3] = (unsigned)(ga >> 32) | (2u << 30);    // global_addr[56:32], type=2
    i32x8 g1 = {};
    g1[0] = (int)(3u << 16);                      // data_size = 8 bytes
    g1[1] = (int)(1024u << 16);                   // tensor_dim0 = 1024 (lo16)
    g1[3] = (int)(1024u << 16);                   // tile_dim0 = 1024
    g1[5] = 1024;                                 // tensor_dim0_stride = 1024
    const i32x4 gz = {};
#if defined(__clang_major__) && __clang_major__ >= 23
    const i32x8 gz8 = {};
    __builtin_amdgcn_tensor_load_to_lds(g0, g1, gz, gz, gz8, 0);
#else
    __builtin_amdgcn_tensor_load_to_lds(g0, g1, gz, gz, 0);
#endif
  }
#else
  const v16h tmp = *(const v16h*)(gsrc + (size_t)tid * 16);
  *(v16h*)(dstLds + (size_t)tid * 16) = tmp;
#endif
  (void)tid; (void)wave;
}

static __device__ __forceinline__ void fa_stage_wait(int wave) {
#if FA_HAVE_TDM
  if (wave == 0) __builtin_amdgcn_s_wait_tensorcnt(0);
#endif
  (void)wave;
}

// ---------------------------------------------------------------------------
// Pass 1: pack K/V fragments (f16) into d_ws, interleaved per KV block.
// Frag order in a block: K[t=0,c=0], K[0,1], K[1,0], K[1,1], V[0..3].
// One wave per (bh, 32-row KV block).
// ---------------------------------------------------------------------------
__launch_bounds__(256, 1)
__global__ void fa_pack_kernel(const float* __restrict__ QKV,
                               _Float16* __restrict__ ws)
{
  const int lane = threadIdx.x & 31;
  const int wave = threadIdx.x >> 5;
  const int lo   = lane & 15;
  const int hi   = lane >> 4;

  const int wid = blockIdx.x * (blockDim.x >> 5) + wave;  // 0..2047 = blk
  const int bh  = wid / ATTN_JB;
  const int jb  = wid % ATTN_JB;
  const int j   = jb * 32;

  const size_t plane = (size_t)ATTN_BH * ATTN_N * ATTN_D;
  const float* __restrict__ Kp = QKV + plane + (size_t)bh * ATTN_N * ATTN_D;
  const float* __restrict__ Vp = QKV + 2 * plane + (size_t)bh * ATTN_N * ATTN_D;

  _Float16* __restrict__ blkBase = ws + (size_t)wid * BLK_HALVES + lane * 16;

  // ---- K fragments: A layout (16x32). lane row = lo,
  //      half jj<8: k = hi*8+jj ; jj>=8: k = 16 + hi*8 + (jj-8) ----
#pragma unroll
  for (int t = 0; t < 2; ++t) {
    const size_t krow = (size_t)(j + t * 16 + lo) * ATTN_D;
#pragma unroll
    for (int c = 0; c < 2; ++c) {
      v16h frag;
#pragma unroll
      for (int g = 0; g < 2; ++g) {
        const v4f a = *(const v4f*)(Kp + krow + c * 32 + g * 16 + hi * 8);
        const v4f b = *(const v4f*)(Kp + krow + c * 32 + g * 16 + hi * 8 + 4);
#pragma unroll
        for (int i = 0; i < 4; ++i) {
          frag[g * 8 + i]     = (_Float16)a[i];
          frag[g * 8 + 4 + i] = (_Float16)b[i];
        }
      }
      *(v16h*)(blkBase + (t * 2 + c) * 512) = frag;
    }
  }

  // ---- V fragments: B layout (32x16). lane col n = lo, e = v*2+h: k = (e&8)*2 + hi*8 + (e&7) ----
#pragma unroll
  for (int c = 0; c < 4; ++c) {
    v16h frag;
#pragma unroll
    for (int v = 0; v < 8; ++v)
#pragma unroll
      for (int h = 0; h < 2; ++h) {
        const int k = ((v * 2 + h) & 8) * 2 + hi * 8 + ((v * 2 + h) & 7);
        frag[v * 2 + h] =
            (_Float16)Vp[(size_t)(j + k) * ATTN_D + c * 16 + lo];
      }
    *(v16h*)(blkBase + (4 + c) * 512) = frag;
  }
}

// ---------------------------------------------------------------------------
// Pass 2: flash attention. Workgroup = 8 waves, one head; KV blocks staged
// through LDS (double-buffered). One wave = two 16-row Q tiles over all KV.
// ---------------------------------------------------------------------------
__launch_bounds__(256, 1)
__global__ void fa_fwd_kernel(const float* __restrict__ QKV,
                              const _Float16* __restrict__ ws,
                              float* __restrict__ Out)
{
  __shared__ _Float16 sbuf[2][BLK_HALVES];   // 2 x 8KB double buffer

  const int tid  = threadIdx.x;
  const int lane = tid & 31;
  const int wave = tid >> 5;
  const int lo   = lane & 15;
  const int hi   = lane >> 4;

  const int wid = blockIdx.x * (blockDim.x >> 5) + wave;  // 0..2047
  const int qsb = ATTN_N / 32;                            // 128
  const int bh  = wid / qsb;   // uniform across the workgroup (128 % 8 == 0)
  const int qs  = wid % qsb;

  const float* __restrict__ Qp = QKV + (size_t)bh * ATTN_N * ATTN_D;
  float* __restrict__ Op = Out + (size_t)bh * ATTN_N * ATTN_D;

  // ---- two Q tiles as B-fragments (loaded once) ----
  v16h qB[2][2];
#pragma unroll
  for (int qt = 0; qt < 2; ++qt) {
    const size_t qrow = (size_t)(qs * 32 + qt * 16 + lo) * ATTN_D;
#pragma unroll
    for (int v = 0; v < 8; ++v)
#pragma unroll
      for (int h = 0; h < 2; ++h) {
        const int d0 = ((v * 2 + h) & 8) * 2 + hi * 8 + ((v * 2 + h) & 7);
        qB[qt][0][v * 2 + h] = (_Float16)Qp[qrow + d0];
        qB[qt][1][v * 2 + h] = (_Float16)Qp[qrow + 32 + d0];
      }
  }

  const v8f vzero = {};
  v8f oAcc[2][4];
#pragma unroll
  for (int qt = 0; qt < 2; ++qt)
#pragma unroll
    for (int c = 0; c < 4; ++c) oAcc[qt][c] = vzero;

  float m_run[2] = {-3.402823466e38f, -3.402823466e38f};
  float l_run[2] = {0.0f, 0.0f};

  const _Float16* __restrict__ gblk =
      ws + (size_t)bh * ATTN_JB * BLK_HALVES;

  // prologue: stage block 0
  fa_stage(&sbuf[0][0], gblk, tid, wave);
  fa_stage_wait(wave);
  __syncthreads();

#pragma unroll 1
  for (int jb = 0; jb < ATTN_JB; ++jb) {
    const int p = jb & 1;
    if (jb + 1 < ATTN_JB)
      fa_stage(&sbuf[p ^ 1][0], gblk + (size_t)(jb + 1) * BLK_HALVES, tid, wave);

    // ---- 8 shared fragments from LDS: K[0..3], V[0..3] ----
    v16h fr[8];
#pragma unroll
    for (int f = 0; f < 8; ++f)
      fr[f] = *(const v16h*)(&sbuf[p][f * 512 + lane * 16]);

#pragma unroll
    for (int qt = 0; qt < 2; ++qt) {
      // ---- S^T tiles: st[t][v] = S[qrow=lo][kv = jb*32 + t*16 + v + 8*hi] ----
      v8f st0 = wmma_f16(fr[0], qB[qt][0], vzero);
      st0     = wmma_f16(fr[1], qB[qt][1], st0);
      v8f st1 = wmma_f16(fr[2], qB[qt][0], vzero);
      st1     = wmma_f16(fr[3], qB[qt][1], st1);

      // ---- online softmax (row = lo, split across half-waves) ----
      float mloc = st0[0];
#pragma unroll
      for (int i = 1; i < 8; ++i) mloc = fmaxf(mloc, st0[i]);
#pragma unroll
      for (int i = 0; i < 8; ++i) mloc = fmaxf(mloc, st1[i]);
      mloc = fmaxf(mloc, __shfl_xor(mloc, 16));
      const float m_new = fmaxf(m_run[qt], mloc);
      const float scale = __expf(m_run[qt] - m_new);

      // P = exp(S - m_new); C-frag of S^T is already P's A-frag layout.
      v16h pA;
      float rs = 0.0f;
#pragma unroll
      for (int i = 0; i < 8; ++i) {
        const float pv = __expf(st0[i] - m_new);
        rs += pv;
        pA[i] = (_Float16)pv;
      }
#pragma unroll
      for (int i = 0; i < 8; ++i) {
        const float pv = __expf(st1[i] - m_new);
        rs += pv;
        pA[8 + i] = (_Float16)pv;
      }
      rs += __shfl_xor(rs, 16);
      l_run[qt] = l_run[qt] * scale + rs;
      m_run[qt] = m_new;

      // ---- rescale O accumulators (C-frag row M = v + 8*hi) ----
#pragma unroll
      for (int v = 0; v < 8; ++v) {
        const float f = __shfl(scale, v + 8 * hi);
#pragma unroll
        for (int c = 0; c < 4; ++c) oAcc[qt][c][v] *= f;
      }

      // ---- O += P(16x32) * V(32x64), four 16-wide d-chunks ----
#pragma unroll
      for (int c = 0; c < 4; ++c)
        oAcc[qt][c] = wmma_f16(pA, fr[4 + c], oAcc[qt][c]);
    }

    fa_stage_wait(wave);   // TDM into buf[p^1] complete before anyone reads it
    __syncthreads();       // also fences this wave's LDS reads of buf[p]
  }

  // ---- normalize and store ----
#pragma unroll
  for (int qt = 0; qt < 2; ++qt) {
    const float inv = 1.0f / l_run[qt];
#pragma unroll
    for (int v = 0; v < 8; ++v) {
      const float f   = __shfl(inv, v + 8 * hi);
      const int   row = qs * 32 + qt * 16 + v + 8 * hi;
#pragma unroll
      for (int c = 0; c < 4; ++c)
        Op[(size_t)row * ATTN_D + c * 16 + lo] = oAcc[qt][c][v] * f;
    }
  }
}

extern "C" void kernel_launch(void* const* d_in, const int* in_sizes, int n_in,
                              void* d_out, int out_size, void* d_ws, size_t ws_size,
                              hipStream_t stream) {
  (void)in_sizes; (void)n_in; (void)out_size; (void)ws_size;
  const float* qkv = (const float*)d_in[0];
  float* out = (float*)d_out;
  _Float16* ws = (_Float16*)d_ws;   // needs 16 MB of packed K/V fragments

  // Pass 1: 2048 waves -> 256 blocks of 8 waves
  fa_pack_kernel<<<dim3(256), dim3(256), 0, stream>>>(qkv, ws);
  // Pass 2: 16 bh * 128 q-superblocks = 2048 waves -> 256 blocks of 8 waves
  fa_fwd_kernel<<<dim3(256), dim3(256), 0, stream>>>(qkv, ws, out);
}
